// LocalShape_12146167513651
// MI455X (gfx1250) — matrix operand order repaired and checked by reading, not compiled, not run on hardware
//
#include <hip/hip_runtime.h>
#include <hip/hip_bf16.h>

// Problem constants (from reference)
#define BATCH   4
#define NPTS    16384
#define NS      4096      // NPOINTS
#define KNN     16        // K
#define P_DIM   64        // planes
#define Q_DIM   128       // shapes
#define BN_EPS  1e-5f

typedef __attribute__((ext_vector_type(16))) _Float16 v16h;
typedef __attribute__((ext_vector_type(8)))  float    v8f;

// ---------------------------------------------------------------------------
// Kernel 1: Farthest Point Sampling. One workgroup (1024 threads = 32 waves)
// per cloud; dist[] + SoA coords resident in LDS (256 KB dynamic, fits the
// 320 KB WGP LDS). Each iteration fuses dist-update with local argmax, then
// wave-shuffle + cross-wave LDS argmax reduction. Writes xyz_new directly.
// ---------------------------------------------------------------------------
__global__ void fps_kernel(const float* __restrict__ xyz,
                           float* __restrict__ xyz_new) {
  extern __shared__ float smem[];
  float* dist = smem;                 // NPTS
  float* px   = smem + NPTS;          // NPTS
  float* py   = px + NPTS;            // NPTS
  float* pz   = py + NPTS;            // NPTS
  __shared__ float rv[32];
  __shared__ int   ri[32];
  __shared__ float s_lx, s_ly, s_lz;

  const int b   = blockIdx.x;
  const int tid = threadIdx.x;
  const float* g = xyz + (size_t)b * NPTS * 3;

  for (int j = tid; j < NPTS; j += 1024) {
    px[j] = g[3 * j + 0];
    py[j] = g[3 * j + 1];
    pz[j] = g[3 * j + 2];
    dist[j] = 1e10f;                  // reference init
  }
  __syncthreads();
  if (tid == 0) {
    s_lx = px[0]; s_ly = py[0]; s_lz = pz[0];   // fps_idx[0] = 0
    float* o = xyz_new + (size_t)b * NS * 3;
    o[0] = px[0]; o[1] = py[0]; o[2] = pz[0];
  }
  __syncthreads();

  for (int i = 1; i < NS; ++i) {
    const float lx = s_lx, ly = s_ly, lz = s_lz;
    float bv = -1.0f; int biv = 0;
    #pragma unroll
    for (int k = 0; k < NPTS / 1024; ++k) {
      const int j = tid + k * 1024;           // strided: conflict-free LDS
      const float dx = px[j] - lx, dy = py[j] - ly, dz = pz[j] - lz;
      const float d  = dx * dx + dy * dy + dz * dz;
      const float nd = fminf(dist[j], d);
      dist[j] = nd;
      if (nd > bv) { bv = nd; biv = j; }      // ascending j => first-max wins
    }
    // wave32 shuffle argmax (tie -> smaller index, matching jnp.argmax)
    #pragma unroll
    for (int off = 16; off > 0; off >>= 1) {
      const float ov = __shfl_down(bv, off, 32);
      const int   oi = __shfl_down(biv, off, 32);
      if (ov > bv || (ov == bv && oi < biv)) { bv = ov; biv = oi; }
    }
    if ((tid & 31) == 0) { rv[tid >> 5] = bv; ri[tid >> 5] = biv; }
    __syncthreads();
    if (tid < 32) {
      bv = rv[tid]; biv = ri[tid];
      #pragma unroll
      for (int off = 16; off > 0; off >>= 1) {
        const float ov = __shfl_down(bv, off, 32);
        const int   oi = __shfl_down(biv, off, 32);
        if (ov > bv || (ov == bv && oi < biv)) { bv = ov; biv = oi; }
      }
      if (tid == 0) {
        const int nxt = biv;
        s_lx = px[nxt]; s_ly = py[nxt]; s_lz = pz[nxt];
        float* o = xyz_new + ((size_t)b * NS + i) * 3;
        o[0] = px[nxt]; o[1] = py[nxt]; o[2] = pz[nxt];
      }
    }
    __syncthreads();
  }
}

// ---------------------------------------------------------------------------
// Kernel 2: kNN (top-16 nearest, ascending distance) per sampled point.
// One thread per query; point cloud streamed through 2048-point LDS tiles.
// Branch-free constant-index insertion keeps the top-16 list in VGPRs.
// ---------------------------------------------------------------------------
#define KTILE 2048
__global__ void knn_kernel(const float* __restrict__ xyz,
                           const float* __restrict__ xyz_new,
                           int* __restrict__ idx_out) {
  __shared__ float tx[KTILE], ty[KTILE], tz[KTILE];
  const int b = blockIdx.x >> 4;                          // 16 blocks / batch
  const int s = ((blockIdx.x & 15) << 8) + threadIdx.x;   // 256 queries/block
  const float* g  = xyz + (size_t)b * NPTS * 3;
  const float* qp = xyz_new + ((size_t)b * NS + s) * 3;
  const float qx = qp[0], qy = qp[1], qz = qp[2];

  float bd[KNN]; int bi[KNN];
  #pragma unroll
  for (int k = 0; k < KNN; ++k) { bd[k] = 3e38f; bi[k] = 0; }

  for (int t0 = 0; t0 < NPTS; t0 += KTILE) {
    __syncthreads();
    for (int j = threadIdx.x; j < KTILE; j += 256) {
      tx[j] = g[3 * (t0 + j) + 0];
      ty[j] = g[3 * (t0 + j) + 1];
      tz[j] = g[3 * (t0 + j) + 2];
    }
    __syncthreads();
    if (t0 + KTILE < NPTS) __builtin_prefetch(&g[3 * (t0 + KTILE)], 0, 1);
    for (int j = 0; j < KTILE; ++j) {
      const float dx = tx[j] - qx, dy = ty[j] - qy, dz = tz[j] - qz;
      const float d  = dx * dx + dy * dy + dz * dz;
      if (d < bd[KNN - 1]) {
        const int id = t0 + j;
        #pragma unroll
        for (int k = KNN - 1; k > 0; --k) {
          if (d < bd[k - 1])      { bd[k] = bd[k - 1]; bi[k] = bi[k - 1]; }
          else if (d < bd[k])     { bd[k] = d;         bi[k] = id;        }
        }
        if (d < bd[0])            { bd[0] = d;         bi[0] = id;        }
      }
    }
  }
  int* o = idx_out + ((size_t)b * NS + s) * KNN;
  #pragma unroll
  for (int k = 0; k < KNN; ++k) o[k] = bi[k];
}

// ---------------------------------------------------------------------------
// Kernel 3: plane features. One thread per (b,s): gather 15 neighbors,
// rel/norm, 64 plane responses dp*|dp|/nrm, max over k. Emits f16 features
// (WMMA A-matrix precision) into workspace.
// ---------------------------------------------------------------------------
__global__ void plane_feat_kernel(const float* __restrict__ xyz,
                                  const float* __restrict__ xyz_new,
                                  const int* __restrict__ idx_out,
                                  const float* __restrict__ w_planes,
                                  _Float16* __restrict__ feat16) {
  __shared__ float wp[P_DIM][3];
  if (threadIdx.x < P_DIM) {
    wp[threadIdx.x][0] = w_planes[threadIdx.x * 3 + 0];
    wp[threadIdx.x][1] = w_planes[threadIdx.x * 3 + 1];
    wp[threadIdx.x][2] = w_planes[threadIdx.x * 3 + 2];
  }
  __syncthreads();

  const int t = blockIdx.x * 256 + threadIdx.x;   // 0..16383
  const int b = t >> 12;
  const int s = t & (NS - 1);
  const float* qp = xyz_new + ((size_t)b * NS + s) * 3;
  const float qx = qp[0], qy = qp[1], qz = qp[2];
  const int* nb = idx_out + ((size_t)b * NS + s) * KNN + 1;  // idx[:,:,1:]

  float acc[P_DIM];
  #pragma unroll
  for (int p = 0; p < P_DIM; ++p) acc[p] = -3e38f;

  for (int k = 0; k < KNN - 1; ++k) {
    const int j = nb[k];
    const float* pp = xyz + ((size_t)b * NPTS + j) * 3;
    const float rx = pp[0] - qx, ry = pp[1] - qy, rz = pp[2] - qz;
    const float nrm = sqrtf(rx * rx + ry * ry + rz * rz) + 1e-8f;
    const float inv = 1.0f / nrm;
    #pragma unroll
    for (int p = 0; p < P_DIM; ++p) {
      const float dp = rx * wp[p][0] + ry * wp[p][1] + rz * wp[p][2];
      const float c  = dp * fabsf(dp) * inv;   // nrm*(dp/nrm)*|dp/nrm|
      acc[p] = fmaxf(acc[p], c);
    }
  }
  _Float16* o = feat16 + (size_t)t * P_DIM;
  #pragma unroll
  for (int p = 0; p < P_DIM; ++p) o[p] = (_Float16)acc[p];
}

// ---------------------------------------------------------------------------
// Kernel 4: shapes GEMM [4096x64]x[64x128] per batch on the WMMA unit
// (v_wmma_f32_16x16x32_f16), fused BN + ReLU, transposed store.
// One wave owns a 16-row s-tile x 128 q columns = 8 C tiles x 2 K-chunks.
// Fragment packing follows ISA 7.12.2 wave32 16-bit layouts.
// ---------------------------------------------------------------------------
__global__ void shapes_wmma_kernel(const _Float16* __restrict__ feat16,
                                   const float* __restrict__ w_shapes,
                                   const float* __restrict__ bn_gamma,
                                   const float* __restrict__ bn_beta,
                                   const float* __restrict__ bn_mean,
                                   const float* __restrict__ bn_var,
                                   float* __restrict__ shapes) {
  const int lane = threadIdx.x & 31;
  const int wave = threadIdx.x >> 5;          // 4 waves / block
  const int b    = blockIdx.x >> 6;           // 64 blocks / batch
  const int stile = ((blockIdx.x & 63) << 2) + wave;
  const int s0    = stile << 4;               // 16 rows of s per wave

  const _Float16* fb = feat16 + (size_t)b * NS * P_DIM;
  const int m  = lane & 15;                   // A row / B column within tile
  const int hi = lane >> 4;                   // lane half selects K sub-range

  v8f c[8];
  #pragma unroll
  for (int qt = 0; qt < 8; ++qt) c[qt] = (v8f){0,0,0,0,0,0,0,0};

  #pragma unroll
  for (int kc = 0; kc < 2; ++kc) {
    const int kb = kc * 32;
    // A fragment: lane<16 -> K kb+0..7 / kb+16..23 ; lane>=16 -> +8 shifted
    const _Float16* fp = fb + (size_t)(s0 + m) * P_DIM + kb + hi * 8;
    v16h a;
    #pragma unroll
    for (int i = 0; i < 8; ++i) { a[i] = fp[i]; a[i + 8] = fp[i + 16]; }

    #pragma unroll
    for (int qt = 0; qt < 8; ++qt) {
      // B fragment: B[K][N] = w_shapes[q][p]; lane -> N, same K order as A (lane-half -> K+8)
      const float* wq = w_shapes + (size_t)(qt * 16 + m) * P_DIM + kb + hi * 8;
      v16h bf;
      #pragma unroll
      for (int i = 0; i < 8; ++i) { bf[i] = (_Float16)wq[i]; bf[i + 8] = (_Float16)wq[i + 16]; }
      c[qt] = __builtin_amdgcn_wmma_f32_16x16x32_f16(
          /*neg_a=*/false, a, /*neg_b=*/false, bf,
          /*c_mod=*/(short)0, c[qt], /*reuse_a=*/false, /*reuse_b=*/false);
    }
  }

  // C/D layout: VGPR r, lanes 0-15: M=r, N=lane; lanes 16-31: M=r+8, N=lane-16
  #pragma unroll
  for (int qt = 0; qt < 8; ++qt) {
    const int q  = qt * 16 + m;
    const float sc = bn_gamma[q] * rsqrtf(bn_var[q] + BN_EPS);
    const float mu = bn_mean[q], be = bn_beta[q];
    float* ob = shapes + ((size_t)b * Q_DIM + q) * NS;   // transposed [b,q,s]
    #pragma unroll
    for (int r = 0; r < 8; ++r) {
      const int srow = s0 + r + hi * 8;
      float v = (c[qt][r] - mu) * sc + be;
      ob[srow] = v > 0.0f ? v : 0.0f;
    }
  }
}

// ---------------------------------------------------------------------------
extern "C" void kernel_launch(void* const* d_in, const int* in_sizes, int n_in,
                              void* d_out, int out_size, void* d_ws, size_t ws_size,
                              hipStream_t stream) {
  (void)in_sizes; (void)n_in; (void)out_size; (void)ws_size;

  const float* xyz      = (const float*)d_in[0];
  const float* w_planes = (const float*)d_in[1];
  const float* w_shapes = (const float*)d_in[2];
  const float* bn_gamma = (const float*)d_in[3];
  const float* bn_beta  = (const float*)d_in[4];
  const float* bn_mean  = (const float*)d_in[5];
  const float* bn_var   = (const float*)d_in[6];

  // Outputs concatenated flat: shapes [4,128,4096] | xyz_new [4,4096,3] | idx [4,4096,16]
  float* out      = (float*)d_out;
  float* shapes   = out;
  float* xyz_new  = out + (size_t)BATCH * Q_DIM * NS;
  int*   idx_out  = (int*)(xyz_new + (size_t)BATCH * NS * 3);

  _Float16* feat16 = (_Float16*)d_ws;     // [4*4096, 64] f16 = 2 MB

  // 256 KB dynamic LDS for FPS (dist + SoA coords); raise the cap once.
  const size_t fps_lds = (size_t)4 * NPTS * sizeof(float);
  hipFuncSetAttribute((const void*)fps_kernel,
                      hipFuncAttributeMaxDynamicSharedMemorySize, (int)fps_lds);

  fps_kernel<<<BATCH, 1024, fps_lds, stream>>>(xyz, xyz_new);
  knn_kernel<<<BATCH * 16, 256, 0, stream>>>(xyz, xyz_new, idx_out);
  plane_feat_kernel<<<(BATCH * NS) / 256, 256, 0, stream>>>(
      xyz, xyz_new, idx_out, w_planes, feat16);
  shapes_wmma_kernel<<<256, 128, 0, stream>>>(
      feat16, w_shapes, bn_gamma, bn_beta, bn_mean, bn_var, shapes);
}
